// ModernVocoder_39986145526082
// MI455X (gfx1250) — matrix operand order repaired and checked by reading, not compiled, not run on hardware
//
#include <hip/hip_runtime.h>
#include <hip/hip_bf16.h>

typedef __attribute__((ext_vector_type(16))) _Float16 v16h;
typedef __attribute__((ext_vector_type(8)))  _Float16 v8h;
typedef __attribute__((ext_vector_type(8)))  float    v8f;

// Problem constants
constexpr int MEL = 80, HID = 128, DILC = 32, NL = 10, NB = 32, T = 4000;
constexpr int SCALE = 200;
constexpr int MELP = 96;                 // MEL padded to multiple of 32 (K dim)

// Fragment buffer layout (in halves). A-fragments stored lane-contiguous:
// block of 512 halves = 32 lanes x 16 halves; lane l reads [blk*512 + l*16 .. +15].
constexpr size_t WN_OFF   = 0;                       // 10 layers x (3 taps x 4 mtiles x 4 ksteps) x 512
constexpr size_t WN_SZ    = (size_t)NL * 48 * 512;   // 245760
constexpr size_t RES_OFF  = WN_OFF + WN_SZ;          // 10 x 8 mtiles x 512
constexpr size_t RES_SZ   = (size_t)NL * 8 * 512;    // 40960
constexpr size_t SKIP_OFF = RES_OFF + RES_SZ;        // 10 ksteps x 8 mtiles x 512 (K=320 stacked)
constexpr size_t SKIP_SZ  = (size_t)80 * 512;        // 40960
constexpr size_t OUT1_OFF = SKIP_OFF + SKIP_SZ;      // 4 ksteps x 8 mtiles x 512
constexpr size_t OUT1_SZ  = (size_t)32 * 512;        // 16384
constexpr size_t IN_OFF   = OUT1_OFF + OUT1_SZ;      // 3 ksteps x 8 mtiles x 512
constexpr size_t IN_SZ    = (size_t)24 * 512;        // 12288
constexpr size_t FRAG_TOTAL = IN_OFF + IN_SZ;        // 356352

__device__ __forceinline__ v8f wmma_f16(v16h a, v16h b, v8f c) {
  return __builtin_amdgcn_wmma_f32_16x16x32_f16(false, a, false, b, (short)0, c,
                                                false, false);
}

// A 16x32 k order used here (contiguous, B matches): row M = lane%16;
// halves j: K = (lane>=16 ? 16 : 0) + j
__device__ __forceinline__ int a_koff(int lane, int j) {
  return ((lane >> 4) << 4) + j;
}

// ---------------------------------------------------------------- prep weights
__global__ __launch_bounds__(128) void prep_weights_kernel(
    const float* __restrict__ wn_w, const float* __restrict__ res_w,
    const float* __restrict__ skip_w, const float* __restrict__ out1_w,
    const float* __restrict__ in_w, const float* __restrict__ skip_b,
    _Float16* __restrict__ frag, float* __restrict__ skipb_sum) {
  const int id = blockIdx.x * 128 + threadIdx.x;
  if (id < (int)(WN_OFF + WN_SZ)) {
    int layer = id / 24576, r = id % 24576;
    int blk = r / 512, lane = (r % 512) / 16, j = r % 16;
    int k = blk >> 4, m = (blk >> 2) & 3, kk = blk & 3;
    int o = m * 16 + (lane & 15);
    int c = kk * 32 + a_koff(lane, j);
    frag[id] = (_Float16)wn_w[((size_t)(layer * 64 + o) * HID + c) * 3 + k];
  } else if (id < (int)(RES_OFF + RES_SZ)) {
    int r = id - (int)RES_OFF;
    int layer = r / 4096, r2 = r % 4096;
    int m = r2 / 512, lane = (r2 % 512) / 16, j = r2 % 16;
    int o = m * 16 + (lane & 15);
    int c = a_koff(lane, j);
    frag[id] = (_Float16)res_w[(size_t)(layer * HID + o) * DILC + c];
  } else if (id < (int)(SKIP_OFF + SKIP_SZ)) {
    int r = id - (int)SKIP_OFF;
    int blk = r / 512, lane = (r % 512) / 16, j = r % 16;
    int kk = blk / 8, m = blk % 8;             // kk == layer index (K = layer*32+c)
    int o = m * 16 + (lane & 15);
    int c = a_koff(lane, j);
    frag[id] = (_Float16)skip_w[(size_t)(kk * HID + o) * DILC + c];
  } else if (id < (int)(OUT1_OFF + OUT1_SZ)) {
    int r = id - (int)OUT1_OFF;
    int blk = r / 512, lane = (r % 512) / 16, j = r % 16;
    int kk = blk / 8, m = blk % 8;
    int o = m * 16 + (lane & 15);
    int c = kk * 32 + a_koff(lane, j);
    frag[id] = (_Float16)out1_w[(size_t)o * HID + c];
  } else if (id < (int)FRAG_TOTAL) {
    int r = id - (int)IN_OFF;
    int blk = r / 512, lane = (r % 512) / 16, j = r % 16;
    int kk = blk / 8, m = blk % 8;
    int o = m * 16 + (lane & 15);
    int c = kk * 32 + a_koff(lane, j);
    frag[id] = (c < MEL) ? (_Float16)in_w[(size_t)o * MEL + c] : (_Float16)0.0f;
  } else {
    int o = id - (int)FRAG_TOTAL;
    if (o < HID) {
      float s = 0.0f;
      for (int i = 0; i < NL; ++i) s += skip_b[i * HID + o];
      skipb_sum[o] = s;
    }
  }
}

// ------------------------------------------- mel (B,80,T) f32 -> (B,T,96) f16
__global__ __launch_bounds__(160) void mel_to_f16_kernel(
    const float* __restrict__ mel, _Float16* __restrict__ mel16) {
  const int b = blockIdx.y;
  const int t = blockIdx.x * 160 + threadIdx.x;
  const float* mb = mel + (size_t)b * MEL * T;
  _Float16 tmp[MELP] __attribute__((aligned(16)));
#pragma unroll
  for (int c = 0; c < MEL; ++c) tmp[c] = (_Float16)mb[(size_t)c * T + t];
#pragma unroll
  for (int c = MEL; c < MELP; ++c) tmp[c] = (_Float16)0.0f;
  _Float16* dst = mel16 + ((size_t)b * T + t) * MELP;
#pragma unroll
  for (int i = 0; i < MELP / 8; ++i)
    *(v8h*)(dst + i * 8) = *(const v8h*)(tmp + i * 8);
}

// ------------------------------------------------ input 1x1 conv + bias + spk
__global__ __launch_bounds__(160) void input_conv_kernel(
    const _Float16* __restrict__ mel16, _Float16* __restrict__ x_out,
    const _Float16* __restrict__ frag, const float* __restrict__ in_b,
    const float* __restrict__ spk_emb, const int* __restrict__ sid_p) {
  const int b = blockIdx.y;
  const int wave = threadIdx.x >> 5, lane = threadIdx.x & 31;
  const int n = lane & 15, hi = lane >> 4;
  const int khalf = hi * 16, mrow = hi * 8;
  const int tcol = blockIdx.x * 80 + wave * 16 + n;
  const int sid = *sid_p;

  const _Float16* mt = mel16 + ((size_t)b * T + tcol) * MELP;
  v8f acc[8];
#pragma unroll
  for (int m = 0; m < 8; ++m) acc[m] = {};
#pragma unroll
  for (int kk = 0; kk < 3; ++kk) {
    v16h bf = *(const v16h*)(mt + kk * 32 + khalf);
#pragma unroll
    for (int m = 0; m < 8; ++m) {
      v16h af = *(const v16h*)(frag + IN_OFF + (size_t)(kk * 8 + m) * 512 + lane * 16);
      acc[m] = wmma_f16(af, bf, acc[m]);
    }
  }
  _Float16* xo = x_out + ((size_t)b * T + tcol) * HID;
  const float* se = spk_emb + (size_t)sid * HID;
#pragma unroll
  for (int m = 0; m < 8; ++m) {
    v8h xn;
#pragma unroll
    for (int r = 0; r < 8; ++r) {
      int o = m * 16 + mrow + r;
      xn[r] = (_Float16)(acc[m][r] + in_b[o] + se[o]);
    }
    *(v8h*)(xo + m * 16 + mrow) = xn;
  }
}

// --------------------------------------- one WaveNet layer (fused conv+gate+res)
__global__ __launch_bounds__(160) void layer_kernel(
    const _Float16* __restrict__ x_in, _Float16* __restrict__ x_out,
    _Float16* __restrict__ gated, const _Float16* __restrict__ frag,
    const float* __restrict__ wn_b, const float* __restrict__ res_b,
    int layer, int dilation) {
  __shared__ _Float16 lds_g[5][16][32] __attribute__((aligned(32)));
  const int b = blockIdx.y;
  const int wave = threadIdx.x >> 5, lane = threadIdx.x & 31;
  const int n = lane & 15, hi = lane >> 4;
  const int khalf = hi * 16, mrow = hi * 8;
  const int tcol = blockIdx.x * 80 + wave * 16 + n;

  const _Float16* xb = x_in + (size_t)b * T * HID;
  const _Float16* wnf = frag + WN_OFF + (size_t)layer * 48 * 512;

  // g = dilated conv, M=64 (4 mtiles), K=3*128
  v8f g[4];
#pragma unroll
  for (int m = 0; m < 4; ++m) g[m] = {};
#pragma unroll
  for (int k = 0; k < 3; ++k) {
    const int t = tcol + (k - 1) * dilation;
    const bool inb = (t >= 0) && (t < T);
#pragma unroll
    for (int kk = 0; kk < 4; ++kk) {
      v16h bf = {};
      if (inb) bf = *(const v16h*)(xb + (size_t)t * HID + kk * 32 + khalf);
#pragma unroll
      for (int m = 0; m < 4; ++m) {
        v16h af = *(const v16h*)(wnf + (size_t)((k * 4 + m) * 4 + kk) * 512 + lane * 16);
        g[m] = wmma_f16(af, bf, g[m]);
      }
    }
  }

  // gated = tanh(g[0:32]) * sigmoid(g[32:64]); store to global + LDS (f16)
#pragma unroll
  for (int mt = 0; mt < 2; ++mt) {
    v8h ga;
#pragma unroll
    for (int r = 0; r < 8; ++r) {
      const int oA = mt * 16 + mrow + r;
      const float aa = g[mt][r] + wn_b[oA];
      const float bb = g[mt + 2][r] + wn_b[oA + 32];
      const float gv = tanhf(aa) * (1.0f / (1.0f + __expf(-bb)));
      ga[r] = (_Float16)gv;
    }
    *(v8h*)&lds_g[wave][n][mt * 16 + mrow] = ga;
    *(v8h*)(gated + ((size_t)b * T + tcol) * (NL * DILC) + layer * DILC + mt * 16 + mrow) = ga;
  }
  __syncthreads();

  // res GEMM (128x32) + x update, B-fragment re-shaped through LDS
  const v16h gf = *(const v16h*)&lds_g[wave][n][khalf];
  const _Float16* resf = frag + RES_OFF + (size_t)layer * 8 * 512;
  _Float16* xo = x_out + ((size_t)b * T + tcol) * HID;
  const _Float16* xoldp = xb + (size_t)tcol * HID;
#pragma unroll
  for (int m = 0; m < 8; ++m) {
    v8f acc = {};
    v16h af = *(const v16h*)(resf + (size_t)m * 512 + lane * 16);
    acc = wmma_f16(af, gf, acc);
    const v8h xold = *(const v8h*)(xoldp + m * 16 + mrow);
    v8h xn;
#pragma unroll
    for (int r = 0; r < 8; ++r) {
      const int o = m * 16 + mrow + r;
      xn[r] = (_Float16)((float)xold[r] + acc[r] + res_b[o]);
    }
    *(v8h*)(xo + m * 16 + mrow) = xn;
  }
}

// ----------- head: skip (K=320 stacked GEMM) -> relu(out1) -> out2 dot -> y(B,T)
__global__ __launch_bounds__(160) void head_kernel(
    const _Float16* __restrict__ gated, const _Float16* __restrict__ frag,
    const float* __restrict__ skipb_sum, const float* __restrict__ out1_b,
    const float* __restrict__ out2_w, const float* __restrict__ out2_b,
    float* __restrict__ y) {
  __shared__ _Float16 lds_s[5][16][HID] __attribute__((aligned(32)));
  const int b = blockIdx.y;
  const int wave = threadIdx.x >> 5, lane = threadIdx.x & 31;
  const int n = lane & 15, hi = lane >> 4;
  const int khalf = hi * 16, mrow = hi * 8;
  const int tcol = blockIdx.x * 80 + wave * 16 + n;

  const _Float16* gt = gated + ((size_t)b * T + tcol) * (NL * DILC);
  v8f s[8];
#pragma unroll
  for (int m = 0; m < 8; ++m) s[m] = {};
#pragma unroll
  for (int kk = 0; kk < NL; ++kk) {
    v16h bf = *(const v16h*)(gt + kk * 32 + khalf);
#pragma unroll
    for (int m = 0; m < 8; ++m) {
      v16h af = *(const v16h*)(frag + SKIP_OFF + (size_t)(kk * 8 + m) * 512 + lane * 16);
      s[m] = wmma_f16(af, bf, s[m]);
    }
  }
#pragma unroll
  for (int m = 0; m < 8; ++m) {
    v8h sh;
#pragma unroll
    for (int r = 0; r < 8; ++r)
      sh[r] = (_Float16)(s[m][r] + skipb_sum[m * 16 + mrow + r]);
    *(v8h*)&lds_s[wave][n][m * 16 + mrow] = sh;
  }
  __syncthreads();

  v8f h[8];
#pragma unroll
  for (int m = 0; m < 8; ++m) h[m] = {};
#pragma unroll
  for (int kk = 0; kk < 4; ++kk) {
    v16h bf = *(const v16h*)&lds_s[wave][n][kk * 32 + khalf];
#pragma unroll
    for (int m = 0; m < 8; ++m) {
      v16h af = *(const v16h*)(frag + OUT1_OFF + (size_t)(kk * 8 + m) * 512 + lane * 16);
      h[m] = wmma_f16(af, bf, h[m]);
    }
  }
  float part = 0.0f;
#pragma unroll
  for (int m = 0; m < 8; ++m) {
#pragma unroll
    for (int r = 0; r < 8; ++r) {
      const int o = m * 16 + mrow + r;
      float v = h[m][r] + out1_b[o];
      v = fmaxf(v, 0.0f);
      part += v * out2_w[o];
    }
  }
  part += __shfl_xor(part, 16, 32);  // lanes l & l^16 hold complementary o halves
  if (lane < 16) y[(size_t)b * T + tcol] = part + out2_b[0];
}

// -------------------------------------------------- linear upsample x200 (BW)
__global__ __launch_bounds__(256) void upsample_kernel(
    const float* __restrict__ y, float* __restrict__ out) {
  const int b = blockIdx.y;
  const int j = blockIdx.x * 256 + threadIdx.x;
  float src = ((float)j + 0.5f) * (1.0f / (float)SCALE) - 0.5f;
  src = fmaxf(src, 0.0f);
  int i0 = (int)src;
  i0 = min(i0, T - 1);
  const int i1 = min(i0 + 1, T - 1);
  const float w = src - (float)i0;
  const float* yb = y + (size_t)b * T;
  out[(size_t)b * (T * SCALE) + j] = yb[i0] * (1.0f - w) + yb[i1] * w;
}

extern "C" void kernel_launch(void* const* d_in, const int* in_sizes, int n_in,
                              void* d_out, int out_size, void* d_ws, size_t ws_size,
                              hipStream_t stream) {
  (void)in_sizes; (void)n_in; (void)out_size; (void)ws_size;
  const float* mel    = (const float*)d_in[0];
  const float* in_w   = (const float*)d_in[1];
  const float* in_b   = (const float*)d_in[2];
  const float* spk    = (const float*)d_in[3];
  const float* wn_w   = (const float*)d_in[4];
  const float* wn_b   = (const float*)d_in[5];
  const float* res_w  = (const float*)d_in[6];
  const float* res_b  = (const float*)d_in[7];
  const float* skip_w = (const float*)d_in[8];
  const float* skip_b = (const float*)d_in[9];
  const float* out1_w = (const float*)d_in[10];
  const float* out1_b = (const float*)d_in[11];
  const float* out2_w = (const float*)d_in[12];
  const float* out2_b = (const float*)d_in[13];
  const int*   sid    = (const int*)d_in[14];
  float* out = (float*)d_out;

  char* ws = (char*)d_ws;
  size_t off = 0;
  auto take = [&](size_t bytes) -> char* {
    char* p = ws + off;
    off += (bytes + 255) & ~(size_t)255;
    return p;
  };
  _Float16* mel16 = (_Float16*)take((size_t)NB * T * MELP * 2);        // 24.6 MB
  _Float16* x0    = (_Float16*)take((size_t)NB * T * HID * 2);         // 32.8 MB
  _Float16* x1    = (_Float16*)take((size_t)NB * T * HID * 2);         // 32.8 MB
  _Float16* gated = (_Float16*)take((size_t)NB * T * NL * DILC * 2);   // 82 MB
  float*    y     = (float*)take((size_t)NB * T * 4);                  // 0.5 MB
  _Float16* frag  = (_Float16*)take(FRAG_TOTAL * 2);
  float*    skb   = (float*)take(HID * 4);

  const int prep_threads = (int)FRAG_TOTAL + HID;  // 356480 = 2785 * 128
  prep_weights_kernel<<<(prep_threads + 127) / 128, 128, 0, stream>>>(
      wn_w, res_w, skip_w, out1_w, in_w, skip_b, frag, skb);

  mel_to_f16_kernel<<<dim3(T / 160, NB), 160, 0, stream>>>(mel, mel16);

  input_conv_kernel<<<dim3(T / 80, NB), 160, 0, stream>>>(mel16, x0, frag,
                                                          in_b, spk, sid);
  _Float16* xin = x0;
  _Float16* xout = x1;
  for (int i = 0; i < NL; ++i) {
    layer_kernel<<<dim3(T / 80, NB), 160, 0, stream>>>(
        xin, xout, gated, frag, wn_b + i * 64, res_b + i * HID, i, 1 << i);
    _Float16* tmp = xin; xin = xout; xout = tmp;
  }

  head_kernel<<<dim3(T / 80, NB), 160, 0, stream>>>(gated, frag, skb, out1_b,
                                                    out2_w, out2_b, y);

  upsample_kernel<<<dim3((T * SCALE) / 256, NB), 256, 0, stream>>>(y, out);
}
